// TwoLayerGCN_47614007443662
// MI455X (gfx1250) — matrix operand order repaired and checked by reading, not compiled, not run on hardware
//
#include <hip/hip_runtime.h>
#include <hip/hip_bf16.h>
#include <stdint.h>

// Two-layer GCN for MI455X (gfx1250, wave32).
// GEMMs via v_wmma_f32_16x16x32_bf16 (bf16 A/B, f32 accumulate).
// Edge aggregation: bf16 gather (halves random-read traffic) + f32 atomics.

#define N_IN   256
#define N_HID  256
#define N_OUT  128
#define BN_EPS 1e-5f

typedef __attribute__((ext_vector_type(16))) __bf16 v16bf;
typedef __attribute__((ext_vector_type(8)))  float  v8f;
typedef __attribute__((ext_vector_type(4)))  unsigned int u32x4;

__device__ __forceinline__ unsigned short f2bf(float f) {
    unsigned int u = __float_as_uint(f);
    unsigned int r = u + 0x7FFFu + ((u >> 16) & 1u);   // round-to-nearest-even
    return (unsigned short)(r >> 16);
}
__device__ __forceinline__ float bf2f(unsigned short s) {
    return __uint_as_float(((unsigned int)s) << 16);
}

// ---------------- elementwise helpers ----------------

__global__ void convert_f32_to_bf16(const float* __restrict__ in,
                                    unsigned short* __restrict__ out, size_t n) {
    size_t i = (size_t)blockIdx.x * blockDim.x + threadIdx.x;
    if (i < n) out[i] = f2bf(in[i]);
}

// Pack W[K x NC] (row-major f32) into per-lane WMMA B layout:
// Wp[((ct*KS + ks)*32 + lane)*16 + e] = bf16( W[(ks*32 + (lane>>4)*16 + e)*NC + ct*16 + (lane&15)] )
__global__ void pack_weight(const float* __restrict__ W, unsigned short* __restrict__ Wp,
                            int K, int NC) {
    size_t i = (size_t)blockIdx.x * blockDim.x + threadIdx.x;
    if (i >= (size_t)K * NC) return;
    int e    = (int)(i & 15);
    int lane = (int)((i >> 4) & 31);
    size_t tk = i >> 9;
    int KS = K / 32;
    int ct = (int)(tk / KS);
    int ks = (int)(tk % KS);
    int col = ct * 16 + (lane & 15);
    int k   = ks * 32 + (lane >> 4) * 16 + e;
    Wp[i] = f2bf(W[(size_t)k * NC + col]);
}

__global__ void degree_kernel(const int* __restrict__ dst, float* __restrict__ deg, int E) {
    int e = blockIdx.x * blockDim.x + threadIdx.x;
    if (e < E) atomicAdd(&deg[dst[e]], 1.0f);
}

__global__ void dinv_kernel(const float* __restrict__ deg, float* __restrict__ dinv, int n) {
    int i = blockIdx.x * blockDim.x + threadIdx.x;
    if (i < n) dinv[i] = rsqrtf(deg[i] + 1.0f);   // +1 self-loop
}

// ---------------- WMMA GEMM: C[nrows x NC] = A[nrows x K] (bf16) @ Bp ----------------
// One wave per 16x16 tile. A row-major bf16, Bp pre-packed per-lane layout.
// Also emits bf16 shadow copy Cb for the downstream random gather.
template<int K, int NC>
__global__ __launch_bounds__(256) void gemm_bf16_wmma(
    const unsigned short* __restrict__ A, const unsigned short* __restrict__ Bp,
    float* __restrict__ C, unsigned short* __restrict__ Cb, int row_tiles)
{
    const int wave = threadIdx.x >> 5;
    const int lane = threadIdx.x & 31;
    const int rt = blockIdx.x * 8 + wave;        // wave-uniform guard (EXEC stays full)
    if (rt >= row_tiles) return;
    const int ct = blockIdx.y;
    const int r  = lane & 15;
    const int hi = lane >> 4;
    constexpr int KS = K / 32;

    v8f acc = {};
    const unsigned short* arow  = A  + (size_t)(rt * 16 + r) * K;
    const unsigned short* bbase = Bp + ((size_t)ct * KS * 32 + lane) * 16;

#pragma unroll
    for (int ks = 0; ks < KS; ++ks) {
        union U { u32x4 q[2]; v16bf v; } a, b;
        const unsigned short* ap = arow + ks * 32 + hi * 16;   // A 16x32 bf16 lane layout
        a.q[0] = *(const u32x4*)(ap);
        a.q[1] = *(const u32x4*)(ap + 8);
        const unsigned short* bp = bbase + (size_t)ks * 32 * 16;
        b.q[0] = *(const u32x4*)(bp);
        b.q[1] = *(const u32x4*)(bp + 8);
        acc = __builtin_amdgcn_wmma_f32_16x16x32_bf16(
            false, a.v, false, b.v, (short)0, acc, false, false);
    }

#pragma unroll
    for (int v = 0; v < 8; ++v) {               // C/D: lane->col, vgpr(+8*hi)->row
        int row = rt * 16 + hi * 8 + v;
        int col = ct * 16 + r;
        size_t idx = (size_t)row * NC + col;
        C[idx]  = acc[v];
        Cb[idx] = f2bf(acc[v]);
    }
}

// ---------------- edge aggregation: agg[dst] += h[src] * dinv[src]*dinv[dst] ----------------
// F/8 threads per edge; each thread gathers 8 bf16 (16B) and issues 8 f32 atomics.
template<int F>
__global__ __launch_bounds__(256) void edge_agg(
    const unsigned short* __restrict__ hb, const int* __restrict__ src,
    const int* __restrict__ dst, const float* __restrict__ dinv,
    float* __restrict__ agg, int E)
{
    constexpr int TPE = F / 8;
    size_t t = (size_t)blockIdx.x * blockDim.x + threadIdx.x;
    size_t e = t / TPE;
    int q = (int)(t % TPE);
    if (e >= (size_t)E) return;
    int s = src[e], d = dst[e];
    float nrm = dinv[s] * dinv[d];
    u32x4 pk = *(const u32x4*)(hb + (size_t)s * F + q * 8);
    float* arow = agg + (size_t)d * F + q * 8;
#pragma unroll
    for (int j = 0; j < 4; ++j) {
        unsigned int w = pk[j];
        atomicAdd(&arow[2 * j],     bf2f((unsigned short)(w & 0xFFFFu)) * nrm);
        atomicAdd(&arow[2 * j + 1], bf2f((unsigned short)(w >> 16))     * nrm);
    }
}

// z = agg + h*dinv^2 + b1 (in-place into agg) and accumulate per-feature sum/sumsq.
__global__ __launch_bounds__(256) void finalize1_stats(
    float* __restrict__ agg, const float* __restrict__ h, const float* __restrict__ dinv,
    const float* __restrict__ b1, float* __restrict__ sum, float* __restrict__ sumsq,
    int n, int rows_per_block)
{
    int f = threadIdx.x;                // 256 threads = 256 features
    int r0 = blockIdx.x * rows_per_block;
    float bias = b1[f];
    float s = 0.f, s2 = 0.f;
    for (int i = 0; i < rows_per_block; ++i) {
        int r = r0 + i;
        if (r >= n) break;
        float di = dinv[r];
        size_t idx = (size_t)r * N_HID + f;
        float z = agg[idx] + h[idx] * di * di + bias;
        agg[idx] = z;
        s += z; s2 += z * z;
    }
    atomicAdd(&sum[f], s);
    atomicAdd(&sumsq[f], s2);
}

__global__ void bn_params(const float* __restrict__ sum, const float* __restrict__ sumsq,
                          const float* __restrict__ gamma, const float* __restrict__ beta,
                          float* __restrict__ bns, float* __restrict__ bnb, int n) {
    int f = threadIdx.x;
    float inv_n = 1.0f / (float)n;
    float mean = sum[f] * inv_n;
    float var  = sumsq[f] * inv_n - mean * mean;   // biased var (training mode)
    float sc   = gamma[f] * rsqrtf(var + BN_EPS);
    bns[f] = sc;
    bnb[f] = beta[f] - mean * sc;
}

// BN + PReLU, write bf16 activations for GEMM2.
__global__ void bn_prelu_bf16(const float* __restrict__ z, const float* __restrict__ bns,
                              const float* __restrict__ bnb, const float* __restrict__ a,
                              unsigned short* __restrict__ out, size_t n) {
    size_t i = (size_t)blockIdx.x * blockDim.x + threadIdx.x;
    if (i >= n) return;
    int f = (int)(i & (N_HID - 1));
    float y = z[i] * bns[f] + bnb[f];
    float a0 = a[0];
    y = (y > 0.f) ? y : a0 * y;
    out[i] = f2bf(y);
}

// out[i] += g2[i]*dinv[row]^2 + b2[f]
__global__ void finalize2(float* __restrict__ out, const float* __restrict__ g2,
                          const float* __restrict__ dinv, const float* __restrict__ b2,
                          size_t n) {
    size_t i = (size_t)blockIdx.x * blockDim.x + threadIdx.x;
    if (i >= n) return;
    int r = (int)(i >> 7);              // N_OUT == 128
    int f = (int)(i & (N_OUT - 1));
    float di = dinv[r];
    out[i] = out[i] + g2[i] * di * di + b2[f];
}

// ---------------- launch ----------------

static inline size_t align256(size_t x) { return (x + 255) & ~(size_t)255; }

extern "C" void kernel_launch(void* const* d_in, const int* in_sizes, int n_in,
                              void* d_out, int out_size, void* d_ws, size_t ws_size,
                              hipStream_t stream) {
    const float* x     = (const float*)d_in[0];
    const int*   ei    = (const int*)d_in[1];       // int32 (harness canonical integer)
    const float* W1    = (const float*)d_in[2];
    const float* b1    = (const float*)d_in[3];
    const float* W2    = (const float*)d_in[4];
    const float* b2    = (const float*)d_in[5];
    const float* gamma = (const float*)d_in[6];
    const float* beta  = (const float*)d_in[7];
    const float* aslp  = (const float*)d_in[8];
    float* out = (float*)d_out;

    const int N = in_sizes[0] / N_IN;               // 100000 (multiple of 16)
    const int E = in_sizes[1] / 2;                  // 1600000
    const int* src = ei;
    const int* dst = ei + E;

    // workspace carve-up (reuse: xb<->hb2, agg1<->g2)
    char* ws = (char*)d_ws;
    size_t off = 0;
    unsigned short* xb   = (unsigned short*)(ws + off); off = align256(off + (size_t)N * N_HID * 2); // xb / hb2
    float*          h1   = (float*)         (ws + off); off = align256(off + (size_t)N * N_HID * 4);
    float*          agg1 = (float*)         (ws + off); off = align256(off + (size_t)N * N_HID * 4); // agg1 / g2
    unsigned short* hb1  = (unsigned short*)(ws + off); off = align256(off + (size_t)N * N_HID * 2);
    unsigned short* g2b  = (unsigned short*)(ws + off); off = align256(off + (size_t)N * N_OUT * 2);
    float*          deg  = (float*)         (ws + off); off = align256(off + (size_t)N * 4);
    float*          dinv = (float*)         (ws + off); off = align256(off + (size_t)N * 4);
    unsigned short* W1p  = (unsigned short*)(ws + off); off = align256(off + (size_t)N_IN * N_HID * 2);
    unsigned short* W2p  = (unsigned short*)(ws + off); off = align256(off + (size_t)N_HID * N_OUT * 2);
    float*          fsum = (float*)         (ws + off); off = align256(off + N_HID * 4);
    float*          fsq  = (float*)         (ws + off); off = align256(off + N_HID * 4);
    float*          bns  = (float*)         (ws + off); off = align256(off + N_HID * 4);
    float*          bnb  = (float*)         (ws + off); off = align256(off + N_HID * 4);
    (void)ws_size; (void)n_in; (void)out_size;

    // zero atomic targets
    hipMemsetAsync(deg,  0, (size_t)N * 4, stream);
    hipMemsetAsync(agg1, 0, (size_t)N * N_HID * 4, stream);
    hipMemsetAsync(fsum, 0, N_HID * 4, stream);
    hipMemsetAsync(fsq,  0, N_HID * 4, stream);
    hipMemsetAsync(out,  0, (size_t)N * N_OUT * 4, stream);

    const int B = 256;
    // x -> bf16
    size_t nx = (size_t)N * N_IN;
    convert_f32_to_bf16<<<(unsigned)((nx + B - 1) / B), B, 0, stream>>>(x, xb, nx);
    // pack weights
    pack_weight<<<(N_IN * N_HID + B - 1) / B, B, 0, stream>>>(W1, W1p, N_IN, N_HID);
    pack_weight<<<(N_HID * N_OUT + B - 1) / B, B, 0, stream>>>(W2, W2p, N_HID, N_OUT);
    // degrees / normalization
    degree_kernel<<<(E + B - 1) / B, B, 0, stream>>>(dst, deg, E);
    dinv_kernel<<<(N + B - 1) / B, B, 0, stream>>>(deg, dinv, N);

    // GEMM1: h1 = x @ W1 (f32 + bf16 shadow)
    const int row_tiles = N / 16;
    dim3 g1((row_tiles + 7) / 8, N_HID / 16);
    gemm_bf16_wmma<N_IN, N_HID><<<g1, 256, 0, stream>>>(xb, W1p, h1, hb1, row_tiles);

    // edge aggregation (layer 1)
    size_t t1 = (size_t)E * (N_HID / 8);
    edge_agg<N_HID><<<(unsigned)((t1 + B - 1) / B), B, 0, stream>>>(hb1, src, dst, dinv, agg1, E);

    // self-loop + bias + BN statistics
    const int nblk = 512;
    const int rpb  = (N + nblk - 1) / nblk;
    finalize1_stats<<<nblk, N_HID, 0, stream>>>(agg1, h1, dinv, b1, fsum, fsq, N, rpb);
    bn_params<<<1, N_HID, 0, stream>>>(fsum, fsq, gamma, beta, bns, bnb, N);

    // BN + PReLU -> bf16 activations (reuse xb)
    size_t nz = (size_t)N * N_HID;
    bn_prelu_bf16<<<(unsigned)((nz + B - 1) / B), B, 0, stream>>>(agg1, bns, bnb, aslp, xb, nz);

    // GEMM2: g2 = act @ W2 (f32 into agg1 region + bf16 shadow)
    dim3 g2grid((row_tiles + 7) / 8, N_OUT / 16);
    gemm_bf16_wmma<N_HID, N_OUT><<<g2grid, 256, 0, stream>>>(xb, W2p, agg1, g2b, row_tiles);

    // edge aggregation (layer 2) directly into d_out
    size_t t2 = (size_t)E * (N_OUT / 8);
    edge_agg<N_OUT><<<(unsigned)((t2 + B - 1) / B), B, 0, stream>>>(g2b, src, dst, dinv, out, E);

    // self-loop + bias
    size_t no = (size_t)N * N_OUT;
    finalize2<<<(unsigned)((no + B - 1) / B), B, 0, stream>>>(out, agg1, dinv, b2, no);
}
